// GAT_42949672960223
// MI455X (gfx1250) — compile-verified, matrix operand order checked by reading, not run on hardware
//
#include <hip/hip_runtime.h>
#include <hip/hip_bf16.h>

typedef __attribute__((ext_vector_type(16))) _Float16 v16h;
typedef __attribute__((ext_vector_type(8)))  float    v8f;

#define NNODES 100000
#define NEDGES 1600000

// ---------- monotone order-encoding for float atomic max via u32 ----------
__device__ __forceinline__ unsigned ordenc(float f) {
    unsigned u = __float_as_uint(f);
    return (u & 0x80000000u) ? ~u : (u | 0x80000000u);
}
__device__ __forceinline__ float orddec(unsigned e) {
    return (e & 0x80000000u) ? __uint_as_float(e & 0x7fffffffu)
                             : __uint_as_float(~e);
}

// ---------- pack W[f32, K=128 x Ncols] into WMMA B-fragment order (f16) ----------
// Fragment order: [ks][ct][lane][half], half h: lane<16 -> K = ks*32 + (h<8 ? h : 8+h)
// lane>=16 -> +8 on the K group (mirrors the ISA 16-bit A-matrix 16x32 layout).
__global__ void pack_w(const float* __restrict__ W, _Float16* __restrict__ Wp,
                       int Ncols, int CT) {
    int t = blockIdx.x * 256 + threadIdx.x;
    int total = 2048 * CT;                 // 4 ksteps * CT * 32 lanes * 16 halves
    if (t >= total) return;
    int h    = t & 15;
    int lane = (t >> 4) & 31;
    int ctks = t >> 9;
    int ct   = ctks % CT;
    int ks   = ctks / CT;
    int n    = ct * 16 + (lane & 15);
    int ko   = (lane & 16) ? 8 : 0;
    int k    = ks * 32 + ko + ((h < 8) ? h : (8 + h));
    Wp[t] = (_Float16)W[(size_t)k * Ncols + n];
}

// ---------- WMMA GEMM: C[M x 16*CT] = A[M x 128] * W, optional bias ----------
template<int CT>
__global__ __launch_bounds__(256) void gemm_wmma(const float* __restrict__ A,
                                                 const _Float16* __restrict__ Wp,
                                                 float* __restrict__ C,
                                                 const float* __restrict__ bias,
                                                 int M) {
    constexpr int NC = CT * 16;
    __shared__ __align__(16) _Float16 Blds[2048 * CT];

    // stage packed weights into LDS (coalesced b128 copies)
    {
        const uint4* s = (const uint4*)Wp;
        uint4*       d = (uint4*)Blds;
        int tid = threadIdx.x;
        #pragma unroll
        for (int i = 0; i < CT; ++i) d[i * 256 + tid] = s[i * 256 + tid];
    }
    __syncthreads();

    const int wave    = threadIdx.x >> 5;
    const int lane    = threadIdx.x & 31;
    const int rowBase = blockIdx.x * 128 + wave * 16;
    int mrow = rowBase + (lane & 15);
    if (mrow >= M) mrow = M - 1;           // clamp loads; stores are guarded
    const int ko = (lane & 16) ? 8 : 0;

    v8f acc[CT];
    #pragma unroll
    for (int ct = 0; ct < CT; ++ct) acc[ct] = (v8f){};

    #pragma unroll
    for (int ks = 0; ks < 4; ++ks) {
        // A fragment: ISA 16-bit A 16x32 layout, built from 4 aligned float4 loads
        const float4* A4 = (const float4*)(A + (size_t)mrow * 128 + ks * 32 + ko);
        float4 f0 = A4[0], f1 = A4[1], f2 = A4[4], f3 = A4[5];
        v16h af;
        af[0]  = (_Float16)f0.x; af[1]  = (_Float16)f0.y;
        af[2]  = (_Float16)f0.z; af[3]  = (_Float16)f0.w;
        af[4]  = (_Float16)f1.x; af[5]  = (_Float16)f1.y;
        af[6]  = (_Float16)f1.z; af[7]  = (_Float16)f1.w;
        af[8]  = (_Float16)f2.x; af[9]  = (_Float16)f2.y;
        af[10] = (_Float16)f2.z; af[11] = (_Float16)f2.w;
        af[12] = (_Float16)f3.x; af[13] = (_Float16)f3.y;
        af[14] = (_Float16)f3.z; af[15] = (_Float16)f3.w;

        #pragma unroll
        for (int ct = 0; ct < CT; ++ct) {
            const v16h bf = *(const v16h*)&Blds[((ks * CT + ct) * 32 + lane) * 16];
            acc[ct] = __builtin_amdgcn_wmma_f32_16x16x32_f16(
                false, af, false, bf, (short)0, acc[ct], false, false);
        }
    }

    // C/D layout: VGPR j -> row (lane<16 ? j : j+8), col = ct*16 + (lane&15)
    #pragma unroll
    for (int ct = 0; ct < CT; ++ct) {
        int col = ct * 16 + (lane & 15);
        float badd = bias ? bias[col] : 0.0f;
        #pragma unroll
        for (int j = 0; j < 8; ++j) {
            int r = rowBase + ((lane & 16) ? (j + 8) : j);
            if (r < M) C[(size_t)r * NC + col] = acc[ct][j] + badd;
        }
    }
}

// ---------- el/er per (node, head) ----------
__global__ void attn_coef(const float* __restrict__ feat,
                          const float* __restrict__ al, const float* __restrict__ ar,
                          float* __restrict__ el, float* __restrict__ er,
                          int Nn, int Hh, int dh) {
    int t = blockIdx.x * blockDim.x + threadIdx.x;
    if (t >= Nn * Hh) return;
    int n = t / Hh, h = t - n * Hh;
    const float4* f4 = (const float4*)(feat + (size_t)n * 128 + h * dh);
    const float4* a4 = (const float4*)(al + h * dh);
    const float4* r4 = (const float4*)(ar + h * dh);
    float sl = 0.f, sr = 0.f;
    for (int i = 0; i < dh / 4; ++i) {
        float4 f = f4[i], a = a4[i], r = r4[i];
        sl += f.x * a.x + f.y * a.y + f.z * a.z + f.w * a.w;
        sr += f.x * r.x + f.y * r.y + f.z * r.z + f.w * r.w;
    }
    el[t] = sl; er[t] = sr;
}

// ---------- segment max over dst (ordered-u32 atomicMax) ----------
__global__ void edge_max(const int* __restrict__ src, const int* __restrict__ dst,
                         const float* __restrict__ el, const float* __restrict__ er,
                         unsigned* __restrict__ menc, int Ee, int Hh) {
    int t = blockIdx.x * blockDim.x + threadIdx.x;
    if (t >= Ee * Hh) return;
    int e = t / Hh, h = t - e * Hh;
    int s_ = src[e], d_ = dst[e];
    float v = el[s_ * Hh + h] + er[d_ * Hh + h];
    v = v > 0.f ? v : 0.2f * v;
    atomicMax(&menc[d_ * Hh + h], ordenc(v));
}

// ---------- exp(e - m[dst]) -> ex, segment sum over dst ----------
__global__ void edge_sum(const int* __restrict__ src, const int* __restrict__ dst,
                         const float* __restrict__ el, const float* __restrict__ er,
                         const unsigned* __restrict__ menc,
                         float* __restrict__ ex, float* __restrict__ ssum,
                         int Ee, int Hh) {
    int t = blockIdx.x * blockDim.x + threadIdx.x;
    if (t >= Ee * Hh) return;
    int e = t / Hh, h = t - e * Hh;
    int s_ = src[e], d_ = dst[e];
    float v = el[s_ * Hh + h] + er[d_ * Hh + h];
    v = v > 0.f ? v : 0.2f * v;
    float m = orddec(menc[d_ * Hh + h]);
    float xv = expf(v - m);
    ex[t] = xv;
    atomicAdd(&ssum[d_ * Hh + h], xv);
}

// ---------- alpha * feat[src] scatter-add into agg[dst] (wave per edge, float4/lane) ----------
__global__ void edge_agg(const int* __restrict__ src, const int* __restrict__ dst,
                         const float* __restrict__ feat, const float* __restrict__ ex,
                         const float* __restrict__ ssum, float* __restrict__ agg,
                         int Ee, int Hh, int log2dh) {
    int t = blockIdx.x * 256 + threadIdx.x;
    if (t >= Ee * 32) return;
    int e = t >> 5, c = t & 31;                 // c: which float4 of the 128-wide row
    int h = c >> (log2dh - 2);                  // dh=32 -> c>>3, dh=128 -> c>>5(=0)
    int s_ = src[e], d_ = dst[e];
    float alpha = ex[(size_t)e * Hh + h] / ssum[d_ * Hh + h];
    float4 f = ((const float4*)feat)[(size_t)s_ * 32 + c];
    float* ap = agg + (size_t)d_ * 128 + c * 4;
    atomicAdd(ap + 0, alpha * f.x);
    atomicAdd(ap + 1, alpha * f.y);
    atomicAdd(ap + 2, alpha * f.z);
    atomicAdd(ap + 3, alpha * f.w);
}

// ---------- per-node: bias + ELU + residual + LayerNorm (wave32 per node) ----------
__global__ __launch_bounds__(256) void node_ln(const float* __restrict__ agg,
                                               const float* __restrict__ bias,
                                               const float* __restrict__ hres,
                                               const float* __restrict__ g,
                                               const float* __restrict__ be,
                                               float* __restrict__ hout, int Nn) {
    int wave = threadIdx.x >> 5, lane = threadIdx.x & 31;
    int n = blockIdx.x * 8 + wave;
    if (n >= Nn) return;
    float4 a  = ((const float4*)agg )[(size_t)n * 32 + lane];
    float4 bb = ((const float4*)bias)[lane];
    float4 r  = ((const float4*)hres)[(size_t)n * 32 + lane];
    float vx = a.x + bb.x; vx = vx > 0.f ? vx : expm1f(vx); vx += r.x;
    float vy = a.y + bb.y; vy = vy > 0.f ? vy : expm1f(vy); vy += r.y;
    float vz = a.z + bb.z; vz = vz > 0.f ? vz : expm1f(vz); vz += r.z;
    float vw = a.w + bb.w; vw = vw > 0.f ? vw : expm1f(vw); vw += r.w;
    float s = vx + vy + vz + vw;
    #pragma unroll
    for (int m = 16; m >= 1; m >>= 1) s += __shfl_xor(s, m, 32);
    float mu = s * (1.0f / 128.0f);
    float dx = vx - mu, dy = vy - mu, dz = vz - mu, dw = vw - mu;
    float vs = dx * dx + dy * dy + dz * dz + dw * dw;
    #pragma unroll
    for (int m = 16; m >= 1; m >>= 1) vs += __shfl_xor(vs, m, 32);
    float inv = rsqrtf(vs * (1.0f / 128.0f) + 1e-5f);
    float4 gg = ((const float4*)g )[lane];
    float4 bt = ((const float4*)be)[lane];
    float4 o;
    o.x = dx * inv * gg.x + bt.x;
    o.y = dy * inv * gg.y + bt.y;
    o.z = dz * inv * gg.z + bt.z;
    o.w = dw * inv * gg.w + bt.w;
    ((float4*)hout)[(size_t)n * 32 + lane] = o;
}

// ---------- layer-2 epilogue: embedding = agg + b2 + residual ----------
__global__ void emb_add(const float* __restrict__ agg, const float* __restrict__ b2,
                        const float* __restrict__ hres, float* __restrict__ emb, int Nn) {
    int t = blockIdx.x * 256 + threadIdx.x;
    if (t >= Nn * 128) return;
    emb[t] = agg[t] + b2[t & 127] + hres[t];
}

extern "C" void kernel_launch(void* const* d_in, const int* in_sizes, int n_in,
                              void* d_out, int out_size, void* d_ws, size_t ws_size,
                              hipStream_t stream) {
    const float* x   = (const float*)d_in[0];
    const int*   src = (const int*)  d_in[1];
    const int*   dst = (const int*)  d_in[2];
    const float* W0  = (const float*)d_in[3];
    const float* al0 = (const float*)d_in[4];
    const float* ar0 = (const float*)d_in[5];
    const float* b0  = (const float*)d_in[6];
    const float* W1  = (const float*)d_in[7];
    const float* al1 = (const float*)d_in[8];
    const float* ar1 = (const float*)d_in[9];
    const float* b1  = (const float*)d_in[10];
    const float* W2  = (const float*)d_in[11];
    const float* al2 = (const float*)d_in[12];
    const float* ar2 = (const float*)d_in[13];
    const float* b2  = (const float*)d_in[14];
    const float* g0  = (const float*)d_in[15];
    const float* be0 = (const float*)d_in[16];
    const float* g1  = (const float*)d_in[17];
    const float* be1 = (const float*)d_in[18];
    const float* Wc  = (const float*)d_in[19];
    const float* bc  = (const float*)d_in[20];

    const int Nn = NNODES, Ee = NEDGES;

    // workspace layout (floats)
    float*    ws    = (float*)d_ws;
    float*    buf0  = ws;                                   // N*128
    float*    buf1  = ws + (size_t)Nn * 128;                // N*128
    float*    buf2  = ws + (size_t)Nn * 128 * 2;            // N*128 (agg)
    float*    el    = ws + (size_t)Nn * 128 * 3;            // N*4
    float*    er    = el + (size_t)Nn * 4;                  // N*4
    unsigned* menc  = (unsigned*)(er + (size_t)Nn * 4);     // N*4
    float*    ssum  = (float*)menc + (size_t)Nn * 4;        // N*4
    float*    ex    = ssum + (size_t)Nn * 4;                // E*4
    _Float16* pack0 = (_Float16*)(ex + (size_t)Ee * 4);
    _Float16* pack1 = pack0 + 16384;
    _Float16* pack2 = pack1 + 16384;
    _Float16* packc = pack2 + 16384;

    float* logits = (float*)d_out;                  // N*64
    float* emb    = (float*)d_out + (size_t)Nn * 64;  // N*128

    // pack weights to f16 fragment order (deterministic; rerun every call)
    pack_w<<<64, 256, 0, stream>>>(W0, pack0, 128, 8);
    pack_w<<<64, 256, 0, stream>>>(W1, pack1, 128, 8);
    pack_w<<<64, 256, 0, stream>>>(W2, pack2, 128, 8);
    pack_w<<<32, 256, 0, stream>>>(Wc, packc, 64, 4);

    const int gGemm  = (Nn + 127) / 128;
    const int gNH4   = (Nn * 4 + 255) / 256;
    const int gNH1   = (Nn + 255) / 256;
    const int gEH4   = (Ee * 4 + 255) / 256;
    const int gEH1   = (Ee + 255) / 256;
    const int gAgg   = (Ee * 32 + 255) / 256;
    const int gNode  = (Nn + 7) / 8;
    const int gElem  = (Nn * 128 + 255) / 256;

    // ---------------- layer 0: x -> buf1 ----------------
    gemm_wmma<8><<<gGemm, 256, 0, stream>>>(x, pack0, buf0, nullptr, Nn);
    attn_coef<<<gNH4, 256, 0, stream>>>(buf0, al0, ar0, el, er, Nn, 4, 32);
    hipMemsetAsync(menc, 0, (size_t)Nn * 4 * sizeof(unsigned), stream);
    hipMemsetAsync(ssum, 0, (size_t)Nn * 4 * sizeof(float), stream);
    hipMemsetAsync(buf2, 0, (size_t)Nn * 128 * sizeof(float), stream);
    edge_max<<<gEH4, 256, 0, stream>>>(src, dst, el, er, menc, Ee, 4);
    edge_sum<<<gEH4, 256, 0, stream>>>(src, dst, el, er, menc, ex, ssum, Ee, 4);
    edge_agg<<<gAgg, 256, 0, stream>>>(src, dst, buf0, ex, ssum, buf2, Ee, 4, 5);
    node_ln<<<gNode, 256, 0, stream>>>(buf2, b0, x, g0, be0, buf1, Nn);

    // ---------------- layer 1: buf1 -> buf0 ----------------
    gemm_wmma<8><<<gGemm, 256, 0, stream>>>(buf1, pack1, buf0, nullptr, Nn);
    attn_coef<<<gNH4, 256, 0, stream>>>(buf0, al1, ar1, el, er, Nn, 4, 32);
    hipMemsetAsync(menc, 0, (size_t)Nn * 4 * sizeof(unsigned), stream);
    hipMemsetAsync(ssum, 0, (size_t)Nn * 4 * sizeof(float), stream);
    hipMemsetAsync(buf2, 0, (size_t)Nn * 128 * sizeof(float), stream);
    edge_max<<<gEH4, 256, 0, stream>>>(src, dst, el, er, menc, Ee, 4);
    edge_sum<<<gEH4, 256, 0, stream>>>(src, dst, el, er, menc, ex, ssum, Ee, 4);
    edge_agg<<<gAgg, 256, 0, stream>>>(src, dst, buf0, ex, ssum, buf2, Ee, 4, 5);
    node_ln<<<gNode, 256, 0, stream>>>(buf2, b1, buf1, g1, be1, buf0, Nn);

    // ---------------- layer 2 (H=1): buf0 -> emb ----------------
    gemm_wmma<8><<<gGemm, 256, 0, stream>>>(buf0, pack2, buf1, nullptr, Nn);
    attn_coef<<<gNH1, 256, 0, stream>>>(buf1, al2, ar2, el, er, Nn, 1, 128);
    hipMemsetAsync(menc, 0, (size_t)Nn * sizeof(unsigned), stream);
    hipMemsetAsync(ssum, 0, (size_t)Nn * sizeof(float), stream);
    hipMemsetAsync(buf2, 0, (size_t)Nn * 128 * sizeof(float), stream);
    edge_max<<<gEH1, 256, 0, stream>>>(src, dst, el, er, menc, Ee, 1);
    edge_sum<<<gEH1, 256, 0, stream>>>(src, dst, el, er, menc, ex, ssum, Ee, 1);
    edge_agg<<<gAgg, 256, 0, stream>>>(src, dst, buf1, ex, ssum, buf2, Ee, 1, 7);
    emb_add<<<gElem, 256, 0, stream>>>(buf2, b2, buf0, emb, Nn);

    // ---------------- classifier: logits = emb @ Wc + bc ----------------
    gemm_wmma<4><<<gGemm, 256, 0, stream>>>(emb, packc, logits, bc, Nn);
}
